// Generator_8349416424056
// MI455X (gfx1250) — matrix operand order repaired and checked by reading, not compiled, not run on hardware
//
#include <hip/hip_runtime.h>
#include <hip/hip_bf16.h>
#include <math.h>

// Problem constants from the reference
#define BB 8
#define SS 64
#define EE 64
#define HH 8
#define DD 8            // head dim = E/H
#define NL 2            // layers
#define VV 50257
#define VPAD 50304      // multiple of 128

typedef __attribute__((ext_vector_type(16))) _Float16 v16h;
typedef __attribute__((ext_vector_type(8)))  _Float16 v8h;
typedef __attribute__((ext_vector_type(8)))  float    v8f;

// ---------------------------------------------------------------------------
// WMMA helpers (CDNA5: V_WMMA_F32_16X16X32_F16, wave32)
// ---------------------------------------------------------------------------
__device__ __forceinline__ v8f wmma32(v16h a, v16h b, v8f c) {
  // 8 args: (neg_a, A, neg_b, B, c_mod, C, reuse_a, reuse_b)
  return __builtin_amdgcn_wmma_f32_16x16x32_f16(false, a, false, b, (short)0, c,
                                                false, false);
}

// ---- A fragment (16x32 f16, row-major, full K): two contiguous v8h loads ----
// Per-lane layout: m = lane%16, elements e0..15 -> k = 16g+e.
__device__ __forceinline__ v16h ldAv(const _Float16* A, int lda, int row0, int k0,
                                     int lane) {
  int g = lane >> 4, m = lane & 15;
  const _Float16* base = A + (row0 + m) * lda + k0 + (g << 4);
  v8h lo = *(const v8h*)(base);
  v8h hi = *(const v8h*)(base + 8);
  v16h f;
#pragma unroll
  for (int e = 0; e < 8; ++e) { f[e] = lo[e]; f[e + 8] = hi[e]; }
  return f;
}

// ---- A fragment, scalar with K zero-padding (used for head-dim K=8) ----
__device__ __forceinline__ v16h ldA(const _Float16* A, int lda, int row0, int k0,
                                    int klim, int lane) {
  int g = lane >> 4, m = lane & 15;
  v16h f;
#pragma unroll
  for (int e = 0; e < 16; ++e) {
    int p = e >> 1, j = e & 1;
    int kr = (g << 4) + e;
    f[e] = (kr < klim) ? A[(row0 + m) * lda + k0 + kr] : (_Float16)0.0f;
  }
  return f;
}

// ---- B fragment from a TRANSPOSED weight matrix Wt[n][k] (contiguous k) ----
// Fragment element e: k = 16*g + e, so one aligned v16h per lane.
__device__ __forceinline__ v16h ldBtv(const _Float16* Wt, int ldk, int col0,
                                      int k0, int lane) {
  int g = lane >> 4, n = lane & 15;
  return *(const v16h*)(Wt + (col0 + n) * ldk + k0 + (g << 4));
}

// ---- B fragment, scalar, row-major B[k][n] with k/n zero-padding ----
__device__ __forceinline__ v16h ldB(const _Float16* Bp, int ldb, int k0, int col0,
                                    int klim, int nlim, int lane) {
  int g = lane >> 4, n = lane & 15;
  v16h f;
#pragma unroll
  for (int e = 0; e < 16; ++e) {
    int kr = (g << 4) + e;
    f[e] = (kr < klim && n < nlim) ? Bp[(k0 + kr) * ldb + col0 + n]
                                   : (_Float16)0.0f;
  }
  return f;
}

// ---- B fragment from transposed source with K zero-padding (scores) ----
__device__ __forceinline__ v16h ldBt(const _Float16* Bp, int ldb, int k0, int col0,
                                     int klim, int lane) {
  int g = lane >> 4, n = lane & 15;
  v16h f;
#pragma unroll
  for (int e = 0; e < 16; ++e) {
    int kr = (g << 4) + e;
    f[e] = (kr < klim) ? Bp[(col0 + n) * ldb + k0 + kr] : (_Float16)0.0f;
  }
  return f;
}

// ---------------------------------------------------------------------------
// Weight conversion: f32 (K x N, nmat matrices of EExEE) -> f16 transposed
// dst[(m*EE + n)*EE + k] = src[(m*EE + k)*EE + n]
// ---------------------------------------------------------------------------
__global__ void tr_weights_kernel(const float* __restrict__ src,
                                  _Float16* __restrict__ dst, int nmat) {
  int i = blockIdx.x * blockDim.x + threadIdx.x;
  int total = nmat * EE * EE;
  if (i < total) {
    int m = i / (EE * EE);
    int k = (i / EE) & 63;
    int n = i & 63;
    dst[(m * EE + n) * EE + k] = (_Float16)src[i];
  }
}

// soft_W (E x V) -> f16 transposed (V x E), rows VV..VPAD-1 left for prep zero
__global__ void tr_softW_kernel(const float* __restrict__ src,
                                _Float16* __restrict__ dst) {
  int i = blockIdx.x * blockDim.x + threadIdx.x;
  if (i < EE * VV) {
    int e = i / VV;
    int v = i - e * VV;
    dst[v * EE + e] = (_Float16)src[i];
  }
}

// ---------------------------------------------------------------------------
// Prep: w = (noise@W0+b0)@W1+b1 -> broadcast kv memory (f16); pos-enc; Y init.
// ---------------------------------------------------------------------------
__global__ void prep_kernel(const float* __restrict__ noise,
                            const float* __restrict__ linW,
                            const float* __restrict__ linb,
                            const float* __restrict__ emb,
                            const int* __restrict__ start_id,
                            float* __restrict__ Y, float* __restrict__ pe,
                            _Float16* __restrict__ kv16,
                            _Float16* __restrict__ feat16,
                            _Float16* __restrict__ softWt) {
  __shared__ float w1[BB * SS];
  __shared__ float w2[BB * SS];
  int t = threadIdx.x;
  // positional encoding: pe[s][2p]=sin(s/10000^(2p/E)), pe[s][2p+1]=cos(...)
  for (int i = t; i < SS * EE; i += 256) {
    int s = i >> 6, e = i & 63;
    int p2 = e & ~1;
    float ang = (float)s * __expf(-logf(10000.0f) * (float)p2 / (float)EE);
    pe[i] = (e & 1) ? cosf(ang) : sinf(ang);
  }
  // w1 = noise @ lin_W[0] + lin_b[0]
  for (int i = t; i < BB * SS; i += 256) {
    int b = i >> 6, s = i & 63;
    float acc = linb[s];
    for (int k = 0; k < SS; ++k) acc += noise[b * SS + k] * linW[k * SS + s];
    w1[i] = acc;
  }
  __syncthreads();
  // w2 = w1 @ lin_W[1] + lin_b[1]
  for (int i = t; i < BB * SS; i += 256) {
    int b = i >> 6, s = i & 63;
    float acc = linb[SS + s];
    for (int k = 0; k < SS; ++k)
      acc += w1[b * SS + k] * linW[SS * SS + k * SS + s];
    w2[i] = acc;
  }
  __syncthreads();
  // kv memory: broadcast w2[b][s] across E (f16)
  for (int i = t; i < BB * SS * EE; i += 256)
    kv16[i] = (_Float16)w2[i >> 6];
  // Y0: zeros, row 0 = emb[start_id] + pe[0]
  int sid = start_id[0];
  for (int i = t; i < BB * SS * EE; i += 256) {
    int rem = i & (SS * EE - 1);
    float v = 0.0f;
    if (rem < EE) v = emb[sid * EE + rem] + pe[rem];
    Y[i] = v;
  }
  // zero the padded A rows (8..15) of the vocab-gemm feature matrix
  for (int i = t; i < 16 * EE; i += 256) feat16[i] = (_Float16)0.0f;
  // zero soft_W pad rows VV..VPAD-1 so edge tiles load clean data
  for (int i = t; i < (VPAD - VV) * EE; i += 256)
    softWt[VV * EE + i] = (_Float16)0.0f;
}

// ---------------------------------------------------------------------------
// Full 2-layer decoder stack for one step. grid = B blocks, 512 thr (16 waves).
// All activations resident in LDS; every GEMM via WMMA with f32 accumulate.
// Weights are f16, TRANSPOSED (Wt[n][k]). Outputs feat16[b] = f16(out[:,tok-1]).
// ---------------------------------------------------------------------------
__global__ __launch_bounds__(512)
void step_layers_kernel(const float* __restrict__ Y,
                        const _Float16* __restrict__ kv16,
                        const _Float16* __restrict__ saWt,
                        const float* __restrict__ sa_b,
                        const _Float16* __restrict__ caWt,
                        const float* __restrict__ ca_b,
                        const _Float16* __restrict__ ff1Wt,
                        const float* __restrict__ ff_b1,
                        const _Float16* __restrict__ ff2Wt,
                        const float* __restrict__ ff_b2,
                        const float* __restrict__ lng,
                        const float* __restrict__ lnb,
                        _Float16* __restrict__ feat16, int tok) {
  extern __shared__ char smem_raw[];
  float*    xs = (float*)smem_raw;           // S*E f32  current activations
  float*    hs = xs + SS * EE;               // S*E f32  residual+block output
  float*    sc = hs + SS * EE;               // S*S f32  per-head scores
  _Float16* af = (_Float16*)(sc + SS * EE);  // S*E f16  x (f16) / attn output
  _Float16* qf = af + SS * EE;               // S*E f16  q  (and FF hidden)
  _Float16* kf = qf + SS * EE;               // S*E f16  k
  _Float16* vf = kf + SS * EE;               // S*E f16  v
  _Float16* pf = vf + SS * EE;               // S*S f16  attention probs

  int tid = threadIdx.x;
  int lane = tid & 31;
  int wave = tid >> 5;  // 0..15
  int b = blockIdx.x;

  for (int i = tid; i < SS * EE; i += 512) xs[i] = Y[b * SS * EE + i];
  __syncthreads();

  for (int ly = 0; ly < NL; ++ly) {
    // ---- two attention blocks: at=0 self, at=1 cross ----
    for (int at = 0; at < 2; ++at) {
      const _Float16* Wt  = (at ? caWt : saWt) + ly * 4 * EE * EE;
      const float*   bias = (at ? ca_b : sa_b) + ly * 4 * EE;
      const _Float16* kvsrc = at ? (kv16 + b * SS * EE) : af;

      for (int i = tid; i < SS * EE; i += 512) af[i] = (_Float16)xs[i];
      __syncthreads();

      // q/k/v projections: 16 tiles of 16x16, one per wave, K=64
      {
        int row0 = (wave >> 2) << 4;
        int col0 = (wave & 3) << 4;
        for (int which = 0; which < 3; ++which) {
          const _Float16* Ain = (which == 0) ? af : kvsrc;
          const _Float16* Wp = Wt + which * EE * EE;
          v8f acc = {};
#pragma unroll
          for (int k0 = 0; k0 < EE; k0 += 32)
            acc = wmma32(ldAv(Ain, EE, row0, k0, lane),
                         ldBtv(Wp, EE, col0, k0, lane), acc);
          _Float16* dst = (which == 0) ? qf : ((which == 1) ? kf : vf);
          int cn = col0 + (lane & 15);
          float bv = bias[which * EE + cn];
          int g = lane >> 4;
#pragma unroll
          for (int r = 0; r < 8; ++r) {
            int m = row0 + r + (g << 3);
            dst[m * EE + cn] = (_Float16)(acc[r] + bv);
          }
        }
      }
      __syncthreads();

      // attention, one head at a time (sc/pf reused)
      for (int h = 0; h < HH; ++h) {
        {  // scores: s = q_h k_h^T / sqrt(d), masked keys -> -1e9
          int qt = wave >> 2, kt = wave & 3;
          v8f acc = {};
          acc = wmma32(ldA(qf, EE, qt * 16, h * DD, DD, lane),
                       ldBt(kf, EE, h * DD, kt * 16, DD, lane), acc);
          int g = lane >> 4;
          int key = kt * 16 + (lane & 15);
#pragma unroll
          for (int r = 0; r < 8; ++r) {
            int q = qt * 16 + r + (g << 3);
            float v = acc[r] * 0.35355339059327373f;  // 1/sqrt(8)
            if (key >= tok) v = -1.0e9f;
            sc[q * SS + key] = v;
          }
        }
        __syncthreads();
        if (tid < SS) {  // softmax per query row
          float mx = -3.0e38f;
          for (int k = 0; k < SS; ++k) mx = fmaxf(mx, sc[tid * SS + k]);
          float sum = 0.0f;
          for (int k = 0; k < SS; ++k) sum += __expf(sc[tid * SS + k] - mx);
          float inv = 1.0f / sum;
          for (int k = 0; k < SS; ++k)
            pf[tid * SS + k] = (_Float16)(__expf(sc[tid * SS + k] - mx) * inv);
        }
        __syncthreads();
        if (wave < 4) {  // o_h = a @ v_h : M=64, N=8 (padded 16), K=64
          int row0 = wave << 4;
          v8f acc = {};
#pragma unroll
          for (int k0 = 0; k0 < SS; k0 += 32)
            acc = wmma32(ldAv(pf, SS, row0, k0, lane),
                         ldB(vf, EE, k0, h * DD, 32, DD, lane), acc);
          int g = lane >> 4, cn = lane & 15;
          if (cn < DD) {
#pragma unroll
            for (int r = 0; r < 8; ++r) {
              int m = row0 + r + (g << 3);
              af[m * EE + h * DD + cn] = (_Float16)acc[r];  // af reused as o
            }
          }
        }
        __syncthreads();
      }

      // out projection + residual -> hs
      {
        int row0 = (wave >> 2) << 4;
        int col0 = (wave & 3) << 4;
        v8f acc = {};
#pragma unroll
        for (int k0 = 0; k0 < EE; k0 += 32)
          acc = wmma32(ldAv(af, EE, row0, k0, lane),
                       ldBtv(Wt + 3 * EE * EE, EE, col0, k0, lane), acc);
        int g = lane >> 4, cn = col0 + (lane & 15);
        float bv = bias[3 * EE + cn];
#pragma unroll
        for (int r = 0; r < 8; ++r) {
          int m = row0 + r + (g << 3);
          hs[m * EE + cn] = acc[r] + bv + xs[m * EE + cn];
        }
      }
      __syncthreads();
      // layernorm (which = at) -> xs
      {
        const float* g_ = lng + (ly * 3 + at) * EE;
        const float* b_ = lnb + (ly * 3 + at) * EE;
        if (tid < SS) {
          float mean = 0.0f;
          for (int e = 0; e < EE; ++e) mean += hs[tid * EE + e];
          mean *= (1.0f / EE);
          float var = 0.0f;
          for (int e = 0; e < EE; ++e) {
            float d = hs[tid * EE + e] - mean;
            var += d * d;
          }
          var *= (1.0f / EE);
          float inv = rsqrtf(var + 1e-5f);
          for (int e = 0; e < EE; ++e)
            xs[tid * EE + e] = (hs[tid * EE + e] - mean) * inv * g_[e] + b_[e];
        }
      }
      __syncthreads();
    }  // at

    // ---- feed-forward ----
    for (int i = tid; i < SS * EE; i += 512) af[i] = (_Float16)xs[i];
    __syncthreads();
    {  // h1 = relu(x @ W1 + b1) -> qf (f16)
      int row0 = (wave >> 2) << 4;
      int col0 = (wave & 3) << 4;
      v8f acc = {};
#pragma unroll
      for (int k0 = 0; k0 < EE; k0 += 32)
        acc = wmma32(ldAv(af, EE, row0, k0, lane),
                     ldBtv(ff1Wt + ly * EE * EE, EE, col0, k0, lane), acc);
      int g = lane >> 4, cn = col0 + (lane & 15);
      float bv = ff_b1[ly * EE + cn];
#pragma unroll
      for (int r = 0; r < 8; ++r) {
        int m = row0 + r + (g << 3);
        qf[m * EE + cn] = (_Float16)fmaxf(acc[r] + bv, 0.0f);
      }
    }
    __syncthreads();
    {  // h2 = h1 @ W2 + b2 + residual -> hs
      int row0 = (wave >> 2) << 4;
      int col0 = (wave & 3) << 4;
      v8f acc = {};
#pragma unroll
      for (int k0 = 0; k0 < EE; k0 += 32)
        acc = wmma32(ldAv(qf, EE, row0, k0, lane),
                     ldBtv(ff2Wt + ly * EE * EE, EE, col0, k0, lane), acc);
      int g = lane >> 4, cn = col0 + (lane & 15);
      float bv = ff_b2[ly * EE + cn];
#pragma unroll
      for (int r = 0; r < 8; ++r) {
        int m = row0 + r + (g << 3);
        hs[m * EE + cn] = acc[r] + bv + xs[m * EE + cn];
      }
    }
    __syncthreads();
    {  // layernorm (which = 2) -> xs
      const float* g_ = lng + (ly * 3 + 2) * EE;
      const float* b_ = lnb + (ly * 3 + 2) * EE;
      if (tid < SS) {
        float mean = 0.0f;
        for (int e = 0; e < EE; ++e) mean += hs[tid * EE + e];
        mean *= (1.0f / EE);
        float var = 0.0f;
        for (int e = 0; e < EE; ++e) {
          float d = hs[tid * EE + e] - mean;
          var += d * d;
        }
        var *= (1.0f / EE);
        float inv = rsqrtf(var + 1e-5f);
        for (int e = 0; e < EE; ++e)
          xs[tid * EE + e] = (hs[tid * EE + e] - mean) * inv * g_[e] + b_[e];
      }
    }
    __syncthreads();
  }  // ly

  if (tid < EE) feat16[b * EE + tid] = (_Float16)xs[(tok - 1) * EE + tid];
}

// ---------------------------------------------------------------------------
// Vocab projection: logits[b][v] = feat[b] . soft_W[:,v] + soft_b[v]
// grid = VPAD/128 blocks, 256 thr (8 waves); each wave one 16x16 WMMA tile.
// softWt is (VPAD x EE) transposed, so B fragments are contiguous b128 loads.
// ---------------------------------------------------------------------------
__global__ void vocab_kernel(const _Float16* __restrict__ feat16,
                             const _Float16* __restrict__ softWt,
                             const float* __restrict__ soft_b,
                             float* __restrict__ logits) {
  int lane = threadIdx.x & 31;
  int wave = threadIdx.x >> 5;
  int col0 = blockIdx.x * 128 + wave * 16;
  // prefetch the next block's slice of soft_W (gfx1250 global_prefetch path)
  __builtin_prefetch(softWt + (size_t)(col0 + 128) * EE, 0, 0);
  v8f acc = {};
#pragma unroll
  for (int k0 = 0; k0 < EE; k0 += 32)
    acc = wmma32(ldAv(feat16, EE, 0, k0, lane),
                 ldBtv(softWt, EE, col0, k0, lane), acc);
  int nlim = VV - col0;
  if (nlim > 16) nlim = 16;
  int g = lane >> 4, nn = lane & 15;
  if (g == 0 && nn < nlim) {  // lanes 0..15 hold batch rows 0..7
    int v = col0 + nn;
    float bv = soft_b[v];
#pragma unroll
    for (int r = 0; r < 8; ++r) logits[r * VPAD + v] = acc[r] + bv;
  }
}

// ---------------------------------------------------------------------------
// Per-batch softmax + argmax + output write + next-token embedding.
// grid = B blocks, 256 threads.
// ---------------------------------------------------------------------------
__global__ void softmax_kernel(const float* __restrict__ logits,
                               const float* __restrict__ emb,
                               const float* __restrict__ pe,
                               float* __restrict__ Y,
                               float* __restrict__ outp, int tok) {
  __shared__ float smax[256];
  __shared__ int sidx[256];
  __shared__ float ssum[256];
  int b = blockIdx.x;
  int t = threadIdx.x;
  const float* lrow = logits + b * VPAD;

  float bm = -3.0e38f;
  int bi = 0;
  for (int v = t; v < VV; v += 256) {
    float x = lrow[v];
    if (x > bm) { bm = x; bi = v; }
  }
  smax[t] = bm;
  sidx[t] = bi;
  __syncthreads();
  for (int s = 128; s > 0; s >>= 1) {
    if (t < s) {
      float a = smax[t], c = smax[t + s];
      int ci = sidx[t + s];
      if (c > a || (c == a && ci < sidx[t])) { smax[t] = c; sidx[t] = ci; }
    }
    __syncthreads();
  }
  float gmax = smax[0];
  int garg = sidx[0];

  float ls = 0.0f;
  for (int v = t; v < VV; v += 256) ls += __expf(lrow[v] - gmax);
  ssum[t] = ls;
  __syncthreads();
  for (int s = 128; s > 0; s >>= 1) {
    if (t < s) ssum[t] += ssum[t + s];
    __syncthreads();
  }
  float inv = 1.0f / ssum[0];

  // output layout (B, S-1, V): step tok writes row tok-1
  float* orow = outp + ((size_t)b * (SS - 1) + (size_t)(tok - 1)) * VV;
  for (int v = t; v < VV; v += 256) orow[v] = __expf(lrow[v] - gmax) * inv;

  // Y[b, tok] = emb[argmax] + pe[tok]
  if (t < EE) Y[(b * SS + tok) * EE + t] = emb[garg * EE + t] + pe[tok * EE + t];
}

// ---------------------------------------------------------------------------
// Host launcher
// ---------------------------------------------------------------------------
extern "C" void kernel_launch(void* const* d_in, const int* in_sizes, int n_in,
                              void* d_out, int out_size, void* d_ws,
                              size_t ws_size, hipStream_t stream) {
  (void)in_sizes; (void)n_in; (void)out_size; (void)ws_size;
  const float* noise  = (const float*)d_in[0];
  const float* lin_W  = (const float*)d_in[1];
  const float* lin_b  = (const float*)d_in[2];
  const float* sa_W   = (const float*)d_in[3];
  const float* sa_b   = (const float*)d_in[4];
  const float* ca_W   = (const float*)d_in[5];
  const float* ca_b   = (const float*)d_in[6];
  const float* ff_W1  = (const float*)d_in[7];
  const float* ff_b1  = (const float*)d_in[8];
  const float* ff_W2  = (const float*)d_in[9];
  const float* ff_b2  = (const float*)d_in[10];
  const float* ln_g   = (const float*)d_in[11];
  const float* ln_b   = (const float*)d_in[12];
  const float* emb    = (const float*)d_in[13];
  const float* soft_W = (const float*)d_in[14];
  const float* soft_b = (const float*)d_in[15];
  const int*   start_id = (const int*)d_in[16];

  char* ws = (char*)d_ws;
  size_t off = 0;
  auto carve = [&](size_t bytes) -> void* {
    void* p = (void*)(ws + off);
    off += (bytes + 255) & ~(size_t)255;
    return p;
  };
  float*    Yb     = (float*)carve((size_t)BB * SS * EE * sizeof(float));
  float*    peb    = (float*)carve((size_t)SS * EE * sizeof(float));
  _Float16* kv16   = (_Float16*)carve((size_t)BB * SS * EE * sizeof(_Float16));
  _Float16* saWt   = (_Float16*)carve((size_t)NL * 4 * EE * EE * sizeof(_Float16));
  _Float16* caWt   = (_Float16*)carve((size_t)NL * 4 * EE * EE * sizeof(_Float16));
  _Float16* ff1Wt  = (_Float16*)carve((size_t)NL * EE * EE * sizeof(_Float16));
  _Float16* ff2Wt  = (_Float16*)carve((size_t)NL * EE * EE * sizeof(_Float16));
  _Float16* softWt = (_Float16*)carve((size_t)VPAD * EE * sizeof(_Float16));
  _Float16* feat16 = (_Float16*)carve((size_t)16 * EE * sizeof(_Float16));
  float*    logits = (float*)carve((size_t)BB * VPAD * sizeof(float));

  const int nSW = EE * VV;  // 3216448
  tr_weights_kernel<<<(NL * 4 * EE * EE + 255) / 256, 256, 0, stream>>>(
      sa_W, saWt, NL * 4);
  tr_weights_kernel<<<(NL * 4 * EE * EE + 255) / 256, 256, 0, stream>>>(
      ca_W, caWt, NL * 4);
  tr_weights_kernel<<<(NL * EE * EE + 255) / 256, 256, 0, stream>>>(ff_W1, ff1Wt,
                                                                    NL);
  tr_weights_kernel<<<(NL * EE * EE + 255) / 256, 256, 0, stream>>>(ff_W2, ff2Wt,
                                                                    NL);
  tr_softW_kernel<<<(nSW + 255) / 256, 256, 0, stream>>>(soft_W, softWt);
  prep_kernel<<<1, 256, 0, stream>>>(noise, lin_W, lin_b, emb, start_id, Yb, peb,
                                     kv16, feat16, softWt);

  size_t shmem = (size_t)(3 * SS * EE) * sizeof(float) +
                 (size_t)(5 * SS * EE) * sizeof(_Float16);  // 90112 B

  for (int tok = 1; tok < SS; ++tok) {
    step_layers_kernel<<<BB, 512, shmem, stream>>>(
        Yb, kv16, saWt, sa_b, caWt, ca_b, ff1Wt, ff_b1, ff2Wt, ff_b2, ln_g, ln_b,
        feat16, tok);
    vocab_kernel<<<VPAD / 128, 256, 0, stream>>>(feat16, softWt, soft_b, logits);
    softmax_kernel<<<BB, 256, 0, stream>>>(logits, emb, peb, Yb, (float*)d_out,
                                           tok);
  }
}
